// E3BaseLineModel3_46729244181055
// MI455X (gfx1250) — matrix operand order repaired and checked by reading, not compiled, not run on hardware
//
#include <hip/hip_runtime.h>
#include <hip/hip_bf16.h>
#include <math.h>

// ---------------------------------------------------------------------------
// MI455X / gfx1250, wave32. Matrix op: v_wmma_f32_16x16x32_f16 (f32 accum).
//
// Per-edge MLP 16->128->256->512->1024 (SiLU between, none after last),
// env linear 1024->96, outer-product expansion with edge_sh (1+3+5 dims).
//
// 64 edges / block. MT=4 m-tiles per wave share each B fragment -> the full
// f16 weight set is read ~once per block from L2 (AI ~64 FLOP/B vs L2).
// NT=1 + k-unroll-2 keeps the wave under 256 VGPRs (no VGPR-MSB juggling)
// while still giving 8 independent WMMAs in flight for hazard cover.
// Weights pre-swizzled to the exact WMMA B-fragment layout, scales folded.
// SiLU = v_exp_f32 + v_rcp_f32.
// ---------------------------------------------------------------------------

typedef _Float16     v16h __attribute__((ext_vector_type(16)));
typedef float        v8f  __attribute__((ext_vector_type(8)));
typedef unsigned int v4u  __attribute__((ext_vector_type(4)));

union Frag { v4u u[2]; v16h h; };

// ---- A fragment: 16x32 f16 (MxK) per ISA layout -----------------------------
// lanes 0-15 : row M=lane,     K = kb*32 + {0..7} and {16..23}
// lanes16-31 : row M=lane-16,  K = kb*32 + {8..15} and {24..31}
__device__ __forceinline__ v16h load_a(const _Float16* act, int strideHalfs,
                                       int mb, int kb, int lane) {
  int row = mb * 16 + (lane & 15);
  int col = kb * 32 + ((lane & 16) ? 8 : 0);
  const v4u* p = (const v4u*)(act + row * strideHalfs + col); // 16B aligned
  Frag f;
  f.u[0] = p[0];   // 8 halfs: K local 0..7
  f.u[1] = p[2];   // 8 halfs: K local 16..23
  return f.h;
}

// ---- B fragment: pre-swizzled in d_ws, 512 halfs (1KB) per (kb,nb) tile ----
__device__ __forceinline__ v16h load_b(const _Float16* W, int numKB,
                                       int nb, int kb, int lane) {
  const v4u* p = (const v4u*)(W + (((size_t)(nb * numKB + kb)) << 9) + (lane << 4));
  Frag f;
  f.u[0] = p[0];
  f.u[1] = p[1];
  return f.h;
}

__device__ __forceinline__ float fast_silu(float v) {
  return v * __builtin_amdgcn_rcpf(1.0f + __expf(-v));   // v_exp + v_rcp
}

// ---------------------------------------------------------------------------
// Generic WMMA layer, M=64 (4 m-tiles). Each wave owns MT m-tiles of one
// n-tile; the B fragment is loaded once and reused across all MT WMMAs.
// MODE 0: SiLU -> f16 LDS    MODE 1: f16 LDS + cutoff*latents NT global store
// MODE 2: f32 LDS (env weights)
// ---------------------------------------------------------------------------
template<int MT, int MODE>
__device__ __forceinline__ void gemm_layer(const _Float16* __restrict__ W,
                                           int numKB, int numNB,
                                           const _Float16* __restrict__ aIn, int sIn,
                                           _Float16* __restrict__ hOut, int sOut,
                                           float* __restrict__ fOut,
                                           const float* __restrict__ cut,
                                           float* __restrict__ latG,
                                           int wave, int lane) {
  const int mGroups = 4 / MT;
  const int groups  = mGroups * numNB;
  for (int t = wave; t < groups; t += 8) {
    const int mg = (t % mGroups) * MT;
    const int nb = t / mGroups;
    v8f c[MT] = {};
#pragma unroll 2
    for (int kb = 0; kb < numKB; ++kb) {
      v16h b = load_b(W, numKB, nb, kb, lane);
      v16h a[MT];
#pragma unroll
      for (int i = 0; i < MT; ++i) a[i] = load_a(aIn, sIn, mg + i, kb, lane);
#pragma unroll
      for (int i = 0; i < MT; ++i)
        c[i] = __builtin_amdgcn_wmma_f32_16x16x32_f16(false, a[i], false, b,
                                                      (short)0, c[i],
                                                      false, false);
    }
    const int mOff = ((lane & 16) ? 8 : 0);
    const int n    = nb * 16 + (lane & 15);
#pragma unroll
    for (int i = 0; i < MT; ++i) {
#pragma unroll
      for (int r = 0; r < 8; ++r) {
        float v = c[i][r];
        const int m = (mg + i) * 16 + mOff + r;
        if (MODE == 0) {
          hOut[m * sOut + n] = (_Float16)fast_silu(v);
        } else if (MODE == 1) {
          hOut[m * sOut + n] = (_Float16)v;                         // env input
          __builtin_nontemporal_store(cut[m] * v,
                                      latG + (size_t)m * 1024 + n); // latents
        } else {
          fOut[m * 96 + n] = v;
        }
      }
    }
  }
}

// ---------------------------------------------------------------------------
// Weight prep: f32 row-major (K x N) -> f16 fragment layout, scale folded,
// K zero-padded to a multiple of 32.
// ---------------------------------------------------------------------------
__global__ void __launch_bounds__(256)
prep_weights(const float* __restrict__ w, _Float16* __restrict__ dst,
             int Kact, int Kpad, int N, float scale) {
  int i = blockIdx.x * 256 + threadIdx.x;
  if (i >= Kpad * N) return;
  int k = i / N, n = i % N;
  int numKB = Kpad >> 5;
  int kb = k >> 5, nb = n >> 4;
  int kl = k & 31;
  int lane = ((kl & 8) ? 16 : 0) + (n & 15);
  size_t off = (((size_t)(nb * numKB + kb)) << 9) + (lane << 4) + ((kl & 7) + ((kl & 16) >> 1));
  float v = (k < Kact) ? w[k * N + n] * scale : 0.0f;
  dst[off] = (_Float16)v;
}

// ---------------------------------------------------------------------------
// Main fused kernel: 64 edges / block, 256 threads = 8 waves.
// Dynamic LDS: bufL 64x1032 halfs (132096B) | bufS 64x520 halfs (66560B) | cut
// (row strides are odd multiples of 16B -> LDS bank spread for ds_load_b128)
// Total ~194KB -> 1 workgroup per 320KB WGP.
// ---------------------------------------------------------------------------
__global__ void __launch_bounds__(256)
e3_edge_mlp(const int* __restrict__ edge_index,
            const float* __restrict__ edge_sh,
            const float* __restrict__ edge_length,
            const float* __restrict__ node_one_hot,
            const float* __restrict__ bessel_freqs,
            const _Float16* __restrict__ ws,
            float* __restrict__ out, int E) {
  extern __shared__ char smem[];
  _Float16* bufL = (_Float16*)smem;                       // 132096 B
  _Float16* bufS = (_Float16*)(smem + 132096);            //  66560 B
  float*    cut  = (float*)(smem + 132096 + 66560);       //    256 B
  float*    wenv = (float*)bufS;                          // reuse after L3

  const int tid  = threadIdx.x;
  const int wave = tid >> 5, lane = tid & 31;
  const int eb   = blockIdx.x * 64;

  // ---- stage 0: bessel + cutoff + one-hot gather -> f16 x (64x32, padded) --
  if (tid < 64) {
    int ge = eb + tid;
    float r  = edge_length[ge];
    float x  = r * 0.2f;                                  // r / R_MAX
    float x2 = x * x, x6 = x2 * x2 * x2, x7 = x6 * x, x8 = x7 * x;
    float co = 1.0f - 28.0f * x6 + 48.0f * x7 - 21.0f * x8;
    co = (x < 1.0f) ? co : 0.0f;
    cut[tid] = co;
    out[(size_t)E * 1312 + ge] = co;                      // cutoff output

    int cidx = edge_index[ge];
    int nidx = edge_index[E + ge];
    _Float16* xr = bufL + tid * 40;
#pragma unroll
    for (int t = 0; t < 4; ++t) {
      xr[t]     = (_Float16)node_one_hot[cidx * 4 + t];
      xr[4 + t] = (_Float16)node_one_hot[nidx * 4 + t];
    }
    float inv_r = __builtin_amdgcn_rcpf(r);               // r >= 0.05
#pragma unroll
    for (int b = 0; b < 8; ++b) {
      float f = bessel_freqs[b];
      xr[8 + b] = (_Float16)(0.4f * __sinf(f * x) * inv_r);  // 2/R*sin(f*r/R)/r
    }
#pragma unroll
    for (int k = 16; k < 32; ++k) xr[k] = (_Float16)0.0f;    // K pad
  }
  __syncthreads();

  float* latG = out + (size_t)eb * 1024;
  gemm_layer<2, 0>(ws + 0,      1,  8,  bufL, 40,  bufS, 136,  nullptr, nullptr, nullptr, wave, lane);
  __syncthreads();
  gemm_layer<4, 0>(ws + 4096,   4,  16, bufS, 136, bufL, 264,  nullptr, nullptr, nullptr, wave, lane);
  __syncthreads();
  gemm_layer<4, 0>(ws + 36864,  8,  32, bufL, 264, bufS, 520,  nullptr, nullptr, nullptr, wave, lane);
  __syncthreads();
  gemm_layer<4, 1>(ws + 167936, 16, 64, bufS, 520, bufL, 1032, nullptr, cut, latG, wave, lane);
  __syncthreads();                                        // bufS reads done
  gemm_layer<2, 2>(ws + 692224, 32, 6,  bufL, 1032, nullptr, 0, wenv, nullptr, nullptr, wave, lane);
  __syncthreads();

  // ---- outer product expansion: features (E x 288) -------------------------
  float* featG = out + (size_t)E * 1024;
  for (int idx = tid; idx < 64 * 288; idx += 256) {
    int e = idx / 288, f = idx % 288;
    int ow, os, w, s;
    if (f < 32)       { ow = 0;  os = 0; w = f;        s = 0;       }
    else if (f < 128) { int rel = f - 32;  ow = 32; os = 1; w = rel / 3; s = rel % 3; }
    else              { int rel = f - 128; ow = 64; os = 4; w = rel / 5; s = rel % 5; }
    float v = wenv[e * 96 + ow + w] * edge_sh[(size_t)(eb + e) * 9 + os + s];
    __builtin_nontemporal_store(v, featG + (size_t)(eb + e) * 288 + f);
  }
}

// ---------------------------------------------------------------------------
extern "C" void kernel_launch(void* const* d_in, const int* in_sizes, int n_in,
                              void* d_out, int out_size, void* d_ws, size_t ws_size,
                              hipStream_t stream) {
  (void)n_in; (void)out_size; (void)ws_size;
  const int*   edge_index   = (const int*)d_in[0];
  // d_in[1] bond_type: unused by the reference
  const float* edge_sh      = (const float*)d_in[2];
  const float* edge_length  = (const float*)d_in[3];
  const float* node_one_hot = (const float*)d_in[4];
  const float* bessel_freqs = (const float*)d_in[5];
  const float* w0           = (const float*)d_in[6];
  const float* w1           = (const float*)d_in[7];
  const float* w2           = (const float*)d_in[8];
  const float* w3           = (const float*)d_in[9];
  const float* w_env        = (const float*)d_in[10];
  const int E = in_sizes[3];                 // edge_length count (160000)

  _Float16* ws  = (_Float16*)d_ws;
  float*    out = (float*)d_out;

  const float SILU = 1.679f;
  // fold norm_from_last / sqrt(fan_in) into f16 weights; pad w0 K: 16 -> 32
  struct { const float* w; size_t off; int Ka, Kp, N; float sc; } P[5] = {
    { w0,        0,   16,   32,  128, 0.25f              },  // 1/sqrt(16)
    { w1,     4096,  128,  128,  256, SILU / 11.3137085f },  // 1.679/sqrt(128)
    { w2,    36864,  256,  256,  512, SILU / 16.0f       },  // 1.679/sqrt(256)
    { w3,   167936,  512,  512, 1024, SILU / 22.6274170f },  // 1.679/sqrt(512)
    { w_env,692224, 1024, 1024,   96, 0.03125f           },  // 1/sqrt(1024)
  };
  for (int i = 0; i < 5; ++i) {
    int total = P[i].Kp * P[i].N;
    hipLaunchKernelGGL(prep_weights, dim3((total + 255) / 256), dim3(256), 0, stream,
                       P[i].w, ws + P[i].off, P[i].Ka, P[i].Kp, P[i].N, P[i].sc);
  }

  const size_t smemBytes = 132096 + 66560 + 256;   // ~194 KB
  hipLaunchKernelGGL(e3_edge_mlp, dim3(E / 64), dim3(256), smemBytes, stream,
                     edge_index, edge_sh, edge_length, node_one_hot, bessel_freqs,
                     ws, out, E);
}
